// Model_74741020885108
// MI455X (gfx1250) — matrix operand order repaired and checked by reading, not compiled, not run on hardware
//
#include <hip/hip_runtime.h>
#include <hip/hip_bf16.h>
#include <math.h>

typedef __attribute__((ext_vector_type(16))) _Float16 v16h;
typedef __attribute__((ext_vector_type(8)))  _Float16 v8h;
typedef __attribute__((ext_vector_type(8)))  float    v8f;

#define BATCH 128
#define SEQ   150
#define EMB   150
#define HID   128
#define G3    384   // 3*HID
#define NCLS  15
#define NFC   2304  // 32*72

// ---------------------------------------------------------------------------
// helpers
// ---------------------------------------------------------------------------
__device__ inline float grp16_sum(float v) {
  v += __shfl_xor(v, 1, 32);
  v += __shfl_xor(v, 2, 32);
  v += __shfl_xor(v, 4, 32);
  v += __shfl_xor(v, 8, 32);
  return v;
}
__device__ inline float sigmoidf(float x) { return 1.f / (1.f + __expf(-x)); }

__device__ inline v16h cat8(v8h lo, v8h hi) {
  return __builtin_shufflevector(lo, hi, 0, 1, 2, 3, 4, 5, 6, 7,
                                 8, 9, 10, 11, 12, 13, 14, 15);
}

// ---------------------------------------------------------------------------
// 0) fp32 -> zero-padded f16 staging (rows x Kp), K <= Kp, Kp % 32 == 0
// ---------------------------------------------------------------------------
__global__ __launch_bounds__(256) void pack_f16(const float* __restrict__ in,
                                                _Float16* __restrict__ out,
                                                int rows, int K, int Kp) {
  int gid = blockIdx.x * 256 + threadIdx.x;
  if (gid >= rows * Kp) return;
  int r = gid / Kp, c = gid % Kp;
  out[gid] = (_Float16)((c < K) ? in[(size_t)r * K + c] : 0.f);
}

// ---------------------------------------------------------------------------
// 1) embedding: c = softmax(code[x], -1); emb[t,e] = sum_{b,w} c*codebook
// ---------------------------------------------------------------------------
__global__ void embed_kernel(const float* __restrict__ code,
                             const float* __restrict__ codebook,
                             const int*   __restrict__ x,
                             float* __restrict__ emb) {
  __shared__ float c_sh[32];
  int t = blockIdx.x;              // 0..B*L-1
  int tid = threadIdx.x;           // 160 threads
  int v = x[t];
  if (tid < 8) {
    const float* row = code + ((size_t)v * 8 + tid) * 4;
    float a0 = row[0], a1 = row[1], a2 = row[2], a3 = row[3];
    float mx = fmaxf(fmaxf(a0, a1), fmaxf(a2, a3));
    float e0 = __expf(a0 - mx), e1 = __expf(a1 - mx);
    float e2 = __expf(a2 - mx), e3 = __expf(a3 - mx);
    float inv = 1.f / (e0 + e1 + e2 + e3);
    c_sh[tid * 4 + 0] = e0 * inv; c_sh[tid * 4 + 1] = e1 * inv;
    c_sh[tid * 4 + 2] = e2 * inv; c_sh[tid * 4 + 3] = e3 * inv;
  }
  __syncthreads();
  if (tid < EMB) {
    float acc = 0.f;
    #pragma unroll
    for (int i = 0; i < 32; ++i) acc += c_sh[i] * codebook[i * EMB + tid];
    emb[(size_t)t * EMB + tid] = acc;
  }
}

// ---------------------------------------------------------------------------
// 2) gi = X_seq · W^T + b  via WMMA f16 16x16x32 (fp32 accum)
//    Xh: f16 (B*L rows in (b,l) order, Kp cols, zero-padded)
//    Wh: f16 (384, Kp).  Y: (L,B,384) with logical row r = l*B + b and the
//    seq reversal folded into the A-row mapping.
//    All operand loads are unconditional contiguous b128s.
// ---------------------------------------------------------------------------
__global__ __launch_bounds__(256) void gi_gemm(const _Float16* __restrict__ Xh,
                                               const _Float16* __restrict__ Wh,
                                               const float* __restrict__ bias,
                                               float* __restrict__ Y,
                                               int L, int Kp, int rev) {
  int wave = threadIdx.x >> 5, lane = threadIdx.x & 31;
  int lhalf = lane >> 4, lmod = lane & 15;
  int tile = blockIdx.x * 8 + wave;
  int Mt = (L * BATCH) / 16;           // 1200
  int nt = tile % 24, mt = tile / 24;
  if (mt >= Mt) return;                // wave-uniform

  int m = mt * 16 + lmod;              // A: lane%16 = M
  int l = m >> 7, b = m & 127;
  int sl = rev ? (L - 1 - l) : l;
  const _Float16* arow = Xh + ((size_t)b * L + sl) * Kp;
  int n = nt * 16 + lmod;              // B: lane%16 = N
  const _Float16* wrow = Wh + (size_t)n * Kp;

  v8f acc = {};
  int nch = Kp >> 5;
  for (int kc = 0; kc < nch; ++kc) {
    int k0 = kc * 32;
    // A 16x32 f16: half h holds K runs [k0+8h, k0+8h+7] and [k0+16+8h, ...+7]
    v8h alo = *(const v8h*)(arow + k0 + lhalf * 8);
    v8h ahi = *(const v8h*)(arow + k0 + 16 + lhalf * 8);
    // B 32x16 f16: half h holds K runs [k0+8h, k0+8h+7] and [k0+16+8h, ...+7]
    v8h blo = *(const v8h*)(wrow + k0 + lhalf * 8);
    v8h bhi = *(const v8h*)(wrow + k0 + 16 + lhalf * 8);
    acc = __builtin_amdgcn_wmma_f32_16x16x32_f16(false, cat8(alo, ahi),
                                                 false, cat8(blo, bhi),
                                                 (short)0, acc, false, false);
  }
  // C/D layout: VGPR i -> M = i + 8*half, N = lane%16
  int ncol = nt * 16 + lmod;
  float bv = bias[ncol];
  #pragma unroll
  for (int i = 0; i < 8; ++i) {
    int mrow = mt * 16 + i + 8 * lhalf;
    Y[(size_t)mrow * G3 + ncol] = acc[i] + bv;
  }
}

// ---------------------------------------------------------------------------
// 3) persistent GRU scan: h in LDS (f16), per-step gh = h·Whh^T via WMMA,
//    fused r/z/n gates.  grid = 2 dirs x 4 batch-chunks of 32 rows.
//    whh is the packed f16 (2,384,128) buffer (L2-resident).
// ---------------------------------------------------------------------------
#define SROWS 32
__global__ __launch_bounds__(256) void gru_scan(const float* __restrict__ gi,      // (2,L,B,384)
                                                const _Float16* __restrict__ whh,  // (2,384,128) f16
                                                const float* __restrict__ b_hh,    // (2,384)
                                                float* __restrict__ hout,          // (B,L,256)
                                                int L) {
  int dir = blockIdx.x >> 2;
  int b0 = (blockIdx.x & 3) * SROWS;
  const float* giD = gi + (size_t)dir * L * BATCH * G3;
  const _Float16* wh = whh + (size_t)dir * G3 * HID;
  const float* bh = b_hh + dir * G3;

  __shared__ __align__(16) _Float16 h_sh[SROWS][136];   // 272B row stride (16B mult)
  __shared__ __align__(16) float    gh_sh[SROWS][392];  // 1568B row stride (16B mult)

  for (int i = threadIdx.x; i < SROWS * HID; i += blockDim.x)
    h_sh[i >> 7][i & 127] = (_Float16)0.f;
  __syncthreads();

  int wave = threadIdx.x >> 5, lane = threadIdx.x & 31;
  int lhalf = lane >> 4, lmod = lane & 15;

  for (int s = 0; s < L; ++s) {
    // ---- gh = h @ wh^T : 48 tiles (2 M-tiles x 24 N-tiles), 8 waves x 6 ----
    for (int t6 = 0; t6 < 6; ++t6) {
      int t = wave * 6 + t6;
      int mt = t & 1, nt = t >> 1;
      v8f acc = {};
      int am = mt * 16 + lmod;
      int n = nt * 16 + lmod;
      const _Float16* wrow = wh + (size_t)n * HID;
      #pragma unroll
      for (int kc = 0; kc < 4; ++kc) {
        int k0 = kc * 32;
        v8h alo = *(const v8h*)&h_sh[am][k0 + lhalf * 8];        // ds_load_b128
        v8h ahi = *(const v8h*)&h_sh[am][k0 + 16 + lhalf * 8];
        v8h blo = *(const v8h*)(wrow + k0 + lhalf * 8);          // global b128
        v8h bhi = *(const v8h*)(wrow + k0 + 16 + lhalf * 8);
        acc = __builtin_amdgcn_wmma_f32_16x16x32_f16(false, cat8(alo, ahi),
                                                     false, cat8(blo, bhi),
                                                     (short)0, acc, false, false);
      }
      int ncol = nt * 16 + lmod;
      #pragma unroll
      for (int i = 0; i < 8; ++i)
        gh_sh[mt * 16 + i + 8 * lhalf][ncol] = acc[i];
    }
    __syncthreads();

    // ---- gates: each thread owns 16 (b,j) cells, all b128 traffic ----
    int bl = threadIdx.x >> 3;          // 0..31
    int jb = (threadIdx.x & 7) * 16;
    int bg = b0 + bl;
    size_t girow = ((size_t)s * BATCH + bg) * G3;
    int l_out = dir ? (L - 1 - s) : s;
    float* orow = hout + ((size_t)bg * L + l_out) * 256 + dir * HID + jb;

    float gir[16], giz[16], gin[16], ghr[16], ghz[16], ghn[16], bhr[16], bhz[16], bhn[16];
    #pragma unroll
    for (int q = 0; q < 4; ++q) {
      *(float4*)&gir[4 * q] = ((const float4*)(giD + girow + jb))[q];
      *(float4*)&giz[4 * q] = ((const float4*)(giD + girow + HID + jb))[q];
      *(float4*)&gin[4 * q] = ((const float4*)(giD + girow + 2 * HID + jb))[q];
      *(float4*)&ghr[4 * q] = *(const float4*)&gh_sh[bl][jb + 4 * q];
      *(float4*)&ghz[4 * q] = *(const float4*)&gh_sh[bl][HID + jb + 4 * q];
      *(float4*)&ghn[4 * q] = *(const float4*)&gh_sh[bl][2 * HID + jb + 4 * q];
      *(float4*)&bhr[4 * q] = ((const float4*)(bh + jb))[q];
      *(float4*)&bhz[4 * q] = ((const float4*)(bh + HID + jb))[q];
      *(float4*)&bhn[4 * q] = ((const float4*)(bh + 2 * HID + jb))[q];
    }
    v8h ho0 = *(const v8h*)&h_sh[bl][jb];
    v8h ho1 = *(const v8h*)&h_sh[bl][jb + 8];
    v8h hn0, hn1;
    float res[16];
    #pragma unroll
    for (int jj = 0; jj < 16; ++jj) {
      float r  = sigmoidf(gir[jj] + ghr[jj] + bhr[jj]);
      float z  = sigmoidf(giz[jj] + ghz[jj] + bhz[jj]);
      float nn = tanhf(gin[jj] + r * (ghn[jj] + bhn[jj]));
      float ho = (jj < 8) ? (float)ho0[jj & 7] : (float)ho1[jj & 7];
      float hn = (1.f - z) * nn + z * ho;
      res[jj] = hn;
      if (jj < 8) hn0[jj & 7] = (_Float16)hn; else hn1[jj & 7] = (_Float16)hn;
    }
    *(v8h*)&h_sh[bl][jb] = hn0;
    *(v8h*)&h_sh[bl][jb + 8] = hn1;
    #pragma unroll
    for (int q = 0; q < 4; ++q)
      ((float4*)orow)[q] = *(const float4*)&res[4 * q];
    __syncthreads();
  }
}

// ---------------------------------------------------------------------------
// 4) out = h2[:,:, :128] + h2[:,:,128:], reshape (B,8,150,16), LN(dim=16)
// ---------------------------------------------------------------------------
__global__ __launch_bounds__(256) void psum_ln0(const float* __restrict__ h2,
                                                const float* __restrict__ g,
                                                const float* __restrict__ bb,
                                                float* __restrict__ out) {
  int gid = blockIdx.x * blockDim.x + threadIdx.x;     // exact: 2457600 threads
  int d = gid & 15;
  int row = gid >> 4;                                  // (b,p,l)
  int l = row % SEQ; int bp = row / SEQ; int p = bp & 7; int b = bp >> 3;
  const float* hrow = h2 + ((size_t)b * SEQ + l) * 256 + p * 16;
  float xv = hrow[d] + hrow[HID + d];
  float mu = grp16_sum(xv) * (1.f / 16.f);
  float dv = xv - mu;
  float var = grp16_sum(dv * dv) * (1.f / 16.f);
  out[gid] = dv * rsqrtf(var + 1e-5f) * g[d] + bb[d];
}

// ---------------------------------------------------------------------------
// 5) caps_conv (uniform routing qk = 1/M) + LN.  in:(B,N,Hin,16) out:(B,M,Hout,16)
//    w: (3,1,N,4,4,M)
// ---------------------------------------------------------------------------
__global__ __launch_bounds__(256) void caps_conv(const float* __restrict__ in,
                                                 const float* __restrict__ w,
                                                 const float* __restrict__ g,
                                                 const float* __restrict__ bb,
                                                 float* __restrict__ out,
                                                 int N, int Hin, int M, int Hout,
                                                 int stride) {
  int gid = blockIdx.x * blockDim.x + threadIdx.x;     // exact multiple of 256
  int dfull = gid & 15; int a = dfull >> 2; int d = dfull & 3;
  int row = gid >> 4;
  int h = row % Hout; int mb = row / Hout; int m = mb % M; int b = mb / M;
  float acc = 0.f;
  for (int k = 0; k < 3; ++k) {
    const float* ibase = in + (((size_t)b * N) * Hin + (h * stride + k)) * 16;
    const float* wbase = w + (size_t)k * N * 16 * M;
    for (int n = 0; n < N; ++n) {
      const float4 ip = *(const float4*)(ibase + (size_t)n * Hin * 16 + a * 4);
      const float* wp = wbase + ((size_t)n * 16 + d) * M + m;  // x stride = 4*M
      acc += ip.x * wp[0] + ip.y * wp[4 * M] + ip.z * wp[8 * M] + ip.w * wp[12 * M];
    }
  }
  acc *= (1.f / (float)M);
  float mu = grp16_sum(acc) * (1.f / 16.f);
  float dv = acc - mu;
  float var = grp16_sum(dv * dv) * (1.f / 16.f);
  out[gid] = dv * rsqrtf(var + 1e-5f) * g[dfull] + bb[dfull];
}

// ---------------------------------------------------------------------------
// 6) caps_fc first pass (uniform qk=1/15) + LN -> v3 (B,15,16)
// ---------------------------------------------------------------------------
__global__ __launch_bounds__(256) void caps_fc_v3(const float* __restrict__ v2flat,
                                                  const float* __restrict__ wfc,
                                                  const float* __restrict__ g,
                                                  const float* __restrict__ bb,
                                                  float* __restrict__ v3) {
  int b = blockIdx.x;
  int t = threadIdx.x;
  if (t >= 240) return;                 // lanes 16..31 of wave 7 idle (group-aligned)
  int dfull = t & 15; int a = dfull >> 2; int d = dfull & 3;
  int m = t >> 4;
  const float* fb = v2flat + (size_t)b * NFC * 16;
  float acc = 0.f;
  for (int n = 0; n < NFC; ++n) {
    const float4 fi = *(const float4*)(fb + (size_t)n * 16 + a * 4);
    const float* wp = wfc + (size_t)n * 240 + d * 15 + m;     // x stride = 60
    acc += fi.x * wp[0] + fi.y * wp[60] + fi.z * wp[120] + fi.w * wp[180];
  }
  acc *= (1.f / (float)NCLS);
  float mu = grp16_sum(acc) * (1.f / 16.f);
  float dv = acc - mu;
  float var = grp16_sum(dv * dv) * (1.f / 16.f);
  v3[(size_t)b * 240 + t] = dv * rsqrtf(var + 1e-5f) * g[dfull] + bb[dfull];
}

// ---------------------------------------------------------------------------
// 7) routing logits[b,n,m] = SCALE * sum_{a,d} (sum_x fi*wfc) * v3
// ---------------------------------------------------------------------------
__global__ __launch_bounds__(256) void fc_logits(const float* __restrict__ v2flat,
                                                 const float* __restrict__ wfc,
                                                 const float* __restrict__ v3,
                                                 float* __restrict__ logits) {
  int gid = blockIdx.x * blockDim.x + threadIdx.x;   // exact: 128*2304*15
  int m = gid % NCLS; int bn = gid / NCLS; int n = bn % NFC; int b = bn / NFC;
  const float* fi = v2flat + ((size_t)b * NFC + n) * 16;
  const float* vb = v3 + ((size_t)b * NCLS + m) * 16;
  const float* wn = wfc + (size_t)n * 240 + m;
  float acc = 0.f;
  #pragma unroll
  for (int a = 0; a < 4; ++a) {
    #pragma unroll
    for (int d = 0; d < 4; ++d) {
      float u = fi[a * 4 + 0] * wn[0 * 60 + d * 15] + fi[a * 4 + 1] * wn[1 * 60 + d * 15]
              + fi[a * 4 + 2] * wn[2 * 60 + d * 15] + fi[a * 4 + 3] * wn[3 * 60 + d * 15];
      acc += u * vb[a * 4 + d];
    }
  }
  logits[gid] = acc * 0.25f;            // SCALE = 1/sqrt(16)
}

// 8) softmax over m (in place) + renormalize by (sum + 1e-10)
__global__ __launch_bounds__(256) void route_softmax(float* __restrict__ logits) {
  int gid = blockIdx.x * blockDim.x + threadIdx.x;   // exact: 128*2304
  float* p = logits + (size_t)gid * NCLS;
  float mx = p[0];
  #pragma unroll
  for (int m = 1; m < NCLS; ++m) mx = fmaxf(mx, p[m]);
  float e[NCLS]; float s = 0.f;
  #pragma unroll
  for (int m = 0; m < NCLS; ++m) { e[m] = __expf(p[m] - mx); s += e[m]; }
  float inv = 1.f / s; float s2 = 0.f;
  #pragma unroll
  for (int m = 0; m < NCLS; ++m) { e[m] *= inv; s2 += e[m]; }
  float inv2 = 1.f / (s2 + 1e-10f);
  #pragma unroll
  for (int m = 0; m < NCLS; ++m) p[m] = e[m] * inv2;
}

// 9) nv3 = sum_n qk*u ; LN ; out[b,m] = nrm^2/(1+nrm^2)
__global__ __launch_bounds__(256) void fc_route_out(const float* __restrict__ v2flat,
                                                    const float* __restrict__ wfc,
                                                    const float* __restrict__ qk,
                                                    const float* __restrict__ g,
                                                    const float* __restrict__ bb,
                                                    float* __restrict__ out) {
  int gid = blockIdx.x * blockDim.x + threadIdx.x;   // exact: 128*15*16 = 30720
  int dfull = gid & 15; int a = dfull >> 2; int d = dfull & 3;
  int row = gid >> 4; int m = row % NCLS; int b = row / NCLS;
  const float* fb = v2flat + (size_t)b * NFC * 16;
  const float* qb = qk + (size_t)b * NFC * NCLS + m;
  float acc = 0.f;
  for (int n = 0; n < NFC; ++n) {
    const float4 fi = *(const float4*)(fb + (size_t)n * 16 + a * 4);
    const float* wp = wfc + (size_t)n * 240 + d * 15 + m;
    float u = fi.x * wp[0] + fi.y * wp[60] + fi.z * wp[120] + fi.w * wp[180];
    acc += qb[(size_t)n * NCLS] * u;
  }
  float mu = grp16_sum(acc) * (1.f / 16.f);
  float dv = acc - mu;
  float var = grp16_sum(dv * dv) * (1.f / 16.f);
  float y = dv * rsqrtf(var + 1e-5f) * g[dfull] + bb[dfull];
  float s2 = grp16_sum(y * y);          // ||nv3||^2
  if (dfull == 0) out[(size_t)b * NCLS + m] = s2 / (1.f + s2);
}

// ---------------------------------------------------------------------------
// launch
// ---------------------------------------------------------------------------
extern "C" void kernel_launch(void* const* d_in, const int* in_sizes, int n_in,
                              void* d_out, int out_size, void* d_ws, size_t ws_size,
                              hipStream_t stream) {
  const float* code     = (const float*)d_in[0];
  const float* codebook = (const float*)d_in[1];
  const float* w_ih_l0  = (const float*)d_in[2];
  const float* w_hh_l0  = (const float*)d_in[3];
  const float* b_ih_l0  = (const float*)d_in[4];
  const float* b_hh_l0  = (const float*)d_in[5];
  const float* w_ih_l1  = (const float*)d_in[6];
  const float* w_hh_l1  = (const float*)d_in[7];
  const float* b_ih_l1  = (const float*)d_in[8];
  const float* b_hh_l1  = (const float*)d_in[9];
  const float* ln0_g = (const float*)d_in[10];
  const float* ln0_b = (const float*)d_in[11];
  const float* w1    = (const float*)d_in[12];
  const float* ln1_g = (const float*)d_in[13];
  const float* ln1_b = (const float*)d_in[14];
  const float* w2    = (const float*)d_in[15];
  const float* ln2_g = (const float*)d_in[16];
  const float* ln2_b = (const float*)d_in[17];
  const float* wfc   = (const float*)d_in[18];
  const float* lnf_g = (const float*)d_in[19];
  const float* lnf_b = (const float*)d_in[20];
  const int*   x     = (const int*)d_in[21];
  float* out = (float*)d_out;
  float* ws  = (float*)d_ws;

  // ---- fp32 workspace layout (floats) ----
  const size_t off_emb = 0;                                         // 2,880,000
  const size_t off_gi  = off_emb + (size_t)BATCH * SEQ * EMB;       // 14,745,600
  const size_t off_h1  = off_gi + 2ull * SEQ * BATCH * G3;          // 4,915,200
  const size_t off_h2  = off_h1 + (size_t)BATCH * SEQ * 256;        // 4,915,200
  const size_t off_f16 = off_h2 + (size_t)BATCH * SEQ * 256;        // f16 region
  // capsule region reuses [0, off_h1)
  const size_t off_ci  = 0;
  const size_t off_v1  = off_ci + (size_t)BATCH * 8 * SEQ * 16;
  const size_t off_v2  = off_v1 + (size_t)BATCH * 32 * 74 * 16;
  const size_t off_v3  = off_v2 + (size_t)BATCH * 32 * 72 * 16;
  const size_t off_lg  = off_v3 + (size_t)BATCH * NCLS * 16;

  float* emb = ws + off_emb;
  float* gi  = ws + off_gi;
  float* h1  = ws + off_h1;
  float* h2  = ws + off_h2;

  // ---- f16 staging sub-layout (units: halves) ----
  _Float16* hb = (_Float16*)(ws + off_f16);
  const size_t T = (size_t)BATCH * SEQ;               // 19200 rows
  _Float16* embh  = hb;                               // 19200 x 160
  _Float16* wih0h = embh  + T * 160;                  // 768 x 160
  _Float16* wih1h = wih0h + 768 * 160;                // 768 x 256
  _Float16* whh0h = wih1h + (size_t)768 * 256;        // 768 x 128
  _Float16* whh1h = whh0h + (size_t)768 * 128;        // 768 x 128
  _Float16* h1h   = whh1h + (size_t)768 * 128;        // 19200 x 256

  const size_t gi_dir = (size_t)SEQ * BATCH * G3;

  // 1) embedding + pack operands
  embed_kernel<<<BATCH * SEQ, 160, 0, stream>>>(code, codebook, x, emb);
  pack_f16<<<(int)(T * 160 / 256), 256, 0, stream>>>(emb, embh, (int)T, EMB, 160);
  pack_f16<<<480, 256, 0, stream>>>(w_ih_l0, wih0h, 768, EMB, 160);
  pack_f16<<<384, 256, 0, stream>>>(w_hh_l0, whh0h, 768, HID, HID);
  pack_f16<<<768, 256, 0, stream>>>(w_ih_l1, wih1h, 768, 256, 256);
  pack_f16<<<384, 256, 0, stream>>>(w_hh_l1, whh1h, 768, HID, HID);

  // 2) GRU layer 0
  gi_gemm<<<3600, 256, 0, stream>>>(embh, wih0h,            b_ih_l0,      gi,          SEQ, 160, 0);
  gi_gemm<<<3600, 256, 0, stream>>>(embh, wih0h + 384 * 160, b_ih_l0 + G3, gi + gi_dir, SEQ, 160, 1);
  gru_scan<<<8, 256, 0, stream>>>(gi, whh0h, b_hh_l0, h1, SEQ);

  // 3) GRU layer 1 (K = 256)
  pack_f16<<<(int)(T * 256 / 256), 256, 0, stream>>>(h1, h1h, (int)T, 256, 256);
  gi_gemm<<<3600, 256, 0, stream>>>(h1h, wih1h,             b_ih_l1,      gi,          SEQ, 256, 0);
  gi_gemm<<<3600, 256, 0, stream>>>(h1h, wih1h + (size_t)384 * 256, b_ih_l1 + G3, gi + gi_dir, SEQ, 256, 1);
  gru_scan<<<8, 256, 0, stream>>>(gi, whh1h, b_hh_l1, h2, SEQ);

  // 4) directional sum + reshape + LN0
  float* caps_in = ws + off_ci;
  float* v1 = ws + off_v1;
  float* v2 = ws + off_v2;
  float* v3 = ws + off_v3;
  float* lg = ws + off_lg;
  psum_ln0<<<9600, 256, 0, stream>>>(h2, ln0_g, ln0_b, caps_in);

  // 5) capsule conv layers (uniform routing passes only — nv1/nv2 are dead)
  caps_conv<<<18944, 256, 0, stream>>>(caps_in, w1, ln1_g, ln1_b, v1, 8, 150, 32, 74, 2);
  caps_conv<<<18432, 256, 0, stream>>>(v1,      w2, ln2_g, ln2_b, v2, 32, 74, 32, 72, 1);

  // 6) capsule fc: uniform pass -> v3, routed pass -> output
  caps_fc_v3<<<BATCH, 256, 0, stream>>>(v2, wfc, lnf_g, lnf_b, v3);
  fc_logits<<<17280, 256, 0, stream>>>(v2, wfc, v3, lg);
  route_softmax<<<1152, 256, 0, stream>>>(lg);
  fc_route_out<<<120, 256, 0, stream>>>(v2, wfc, lg, lnf_g, lnf_b, out);
}
